// Separate_non_local_59373627900156
// MI455X (gfx1250) — compile-verified, matrix operand order checked by reading, not run on hardware
//
#include <hip/hip_runtime.h>

typedef __attribute__((ext_vector_type(16))) _Float16 v16h;
typedef __attribute__((ext_vector_type(8)))  _Float16 v8h;
typedef __attribute__((ext_vector_type(8)))  float    v8f;
typedef __attribute__((ext_vector_type(2)))  float    v2f;

#define B_   2
#define C_   128
#define HW_  4096
#define N_   8192
#define KT   32
#define NWAVE 4
#define NBLK (N_ / KT)   // 256 key blocks

// --- CDNA5 async global->LDS copy (16B per lane), tracked by ASYNCcnt ---
__device__ __forceinline__ unsigned lds_off(const void* p) {
  // flat shared pointer: low 32 bits == wave-relative LDS byte address
  return (unsigned)(unsigned long long)p;
}
__device__ __forceinline__ void async_ld16(const void* g, unsigned lds) {
  asm volatile("global_load_async_to_lds_b128 %0, %1, off"
               :: "v"(lds), "v"((unsigned long long)g) : "memory");
}
__device__ __forceinline__ void wait_async8() {
  asm volatile("s_wait_asynccnt 0x8" ::: "memory");
}
__device__ __forceinline__ void wait_async0() {
  asm volatile("s_wait_asynccnt 0x0" ::: "memory");
}

__device__ __forceinline__ v16h pack16(v8h lo, v8h hi) {
  v16h r;
  #pragma unroll
  for (int i = 0; i < 8; ++i) { r[i] = lo[i]; r[i + 8] = hi[i]; }
  return r;
}

// 16x16 f32 GEMM tile over K=128 in steps of 4 on the f32 matrix pipe.
// arow: token-major LDS row (A, row = lane&15); brow: weight row (B, col = lane&15).
__device__ __forceinline__ v8f gemm128_f32(const float* __restrict__ brow,
                                           const float* arow, float bias,
                                           int hi) {
  v8f acc;
  #pragma unroll
  for (int i = 0; i < 8; ++i) acc[i] = bias;
  #pragma unroll 4
  for (int ks = 0; ks < 32; ++ks) {
    const int cb = ks * 4 + hi * 2;
    v2f a  = *(const v2f*)(arow + cb);
    v2f bw = *(const v2f*)(brow + cb);
    acc = __builtin_amdgcn_wmma_f32_16x16x4_f32(false, a, false, bw,
              (short)0, acc, false, false);
  }
  return acc;
}

// ================= Kernel 1: fused QKV projection via f32 WMMA ==============
// Per block: 32 tokens x 128 outputs x 3 matrices, tiles of 16x16, K-steps of 4
// on V_WMMA_F32_16X16X4_F32 (full f32 precision for the projections).
// Qh/Kh: [b][n][c] f16 (token-major). Vt: [b][c][n] f16 (channel-major).
__global__ __launch_bounds__(256) void snl_qkv(
    const float* __restrict__ ref, const float* __restrict__ nei,
    const float* __restrict__ wA, const float* __restrict__ bA,
    const float* __restrict__ wB, const float* __restrict__ bB,
    const float* __restrict__ wD, const float* __restrict__ bD,
    _Float16* __restrict__ Qh, _Float16* __restrict__ Kh,
    _Float16* __restrict__ Vt)
{
  __shared__ float xs[32][C_];   // [token][channel], 16 KB
  const int tid = threadIdx.x;
  const int b   = blockIdx.x / (N_ / 32);
  const int n0  = (blockIdx.x % (N_ / 32)) * 32;
  const float* xb = (n0 < HW_) ? ref : nei;
  const int   off = (n0 < HW_) ? n0 : n0 - HW_;
  {
    // transpose-load x tile: channel c = tid/2, 16 tokens per thread
    int c = tid >> 1, part = (tid & 1) * 16;
    const float* src = xb + ((size_t)(b * C_ + c)) * HW_ + off + part;
    #pragma unroll
    for (int j = 0; j < 16; ++j) xs[part + j][c] = src[j];
  }
  __syncthreads();

  const int wave = tid >> 5, lane = tid & 31;
  const int lo16 = lane & 15, hi = lane >> 4;

  for (int job = wave; job < 48; job += 8) {      // job index is wave-uniform
    const int rem = job & 15;
    const int t0  = (rem >> 3) * 16;              // token tile base
    const int og  = (rem & 7) * 16;               // out-channel tile base
    const float* arow = &xs[t0 + lo16][0];
    const size_t wofs = (size_t)(og + lo16) * C_;

    if (job < 16) {                               // ---- Q (token-major) ----
      v8f acc = gemm128_f32(wA + wofs, arow, bA[og + lo16], hi);
      #pragma unroll
      for (int v = 0; v < 8; ++v) {
        const int r = v + 8 * hi;
        Qh[((size_t)b * N_ + n0 + t0 + r) * C_ + og + lo16] = (_Float16)acc[v];
      }
    } else if (job < 32) {                        // ---- K (token-major) ----
      v8f acc = gemm128_f32(wB + wofs, arow, bB[og + lo16], hi);
      #pragma unroll
      for (int v = 0; v < 8; ++v) {
        const int r = v + 8 * hi;
        Kh[((size_t)b * N_ + n0 + t0 + r) * C_ + og + lo16] = (_Float16)acc[v];
      }
    } else {                                      // ---- V (channel-major) ----
      v8f acc = gemm128_f32(wD + wofs, arow, bD[og + lo16], hi);
      v8h pk;                                     // 8 consecutive tokens/lane
      #pragma unroll
      for (int i = 0; i < 8; ++i) pk[i] = (_Float16)acc[i];
      *(v8h*)(Vt + ((size_t)(b * C_ + og + lo16)) * N_ + n0 + t0 + 8 * hi) = pk;
    }
  }
}

// ================= Kernel 2: flash attention with WMMA ======================
// 128 threads = 4 waves; wave owns 16 query rows. E: [b][n][c] f16.
// Row-sum of softmax numerators is fused into a 9th WMMA tile (P x ones).
__global__ __launch_bounds__(128) void snl_attn(
    const _Float16* __restrict__ Qh, const _Float16* __restrict__ Kh,
    const _Float16* __restrict__ Vt, _Float16* __restrict__ Eh)
{
  __shared__ _Float16 Kls[2][KT][C_];      // key block, token-major
  __shared__ _Float16 Vls[2][C_][KT];      // value block, channel-major
  __shared__ _Float16 Pls[NWAVE][16][KT];  // per-wave P transpose scratch

  const int tid  = threadIdx.x;
  const int wave = tid >> 5;
  const int lane = tid & 31;
  const int lo16 = lane & 15;
  const int hi   = lane >> 4;
  const int b     = blockIdx.x >> 7;
  const int qbase = (blockIdx.x & 127) * 64;

  // Q fragments for this wave's 16 rows (held in registers for whole kernel)
  const _Float16* qrow = Qh + ((size_t)b * N_ + qbase + wave * 16 + lo16) * C_;
  v16h qa[4];
  #pragma unroll
  for (int ks = 0; ks < 4; ++ks) {
    const int kb = ks * 32;
    v8h c0 = *(const v8h*)(qrow + kb + hi * 8);
    v8h c1 = *(const v8h*)(qrow + kb + 16 + hi * 8);
    qa[ks] = pack16(c0, c1);
  }

  const v8f vzero = {0.f, 0.f, 0.f, 0.f, 0.f, 0.f, 0.f, 0.f};
  v16h ones;
  #pragma unroll
  for (int i = 0; i < 16; ++i) ones[i] = (_Float16)1.0f;

  v8f acc[8];
  v8f accl = vzero;          // row-sum accumulator tile (P x ones)
  float mrow[8];
  #pragma unroll
  for (int i = 0; i < 8; ++i) { acc[i] = vzero; mrow[i] = -1e30f; }

  const char* kbase_g = (const char*)(Kh + ((size_t)b * N_) * C_);
  auto issue = [&](int buf, int m0) {
    // K block: contiguous 8 KB; 128 threads x 4 x 16B
    const char* ksrc = kbase_g + (size_t)m0 * C_ * 2;
    char* kdst = (char*)&Kls[buf][0][0];
    #pragma unroll
    for (int i = 0; i < 4; ++i) {
      const int o = tid * 16 + i * 2048;
      async_ld16(ksrc + o, lds_off(kdst + o));
    }
    // V block: one 64B channel row per thread
    const _Float16* vsrc = Vt + ((size_t)(b * C_ + tid)) * N_ + m0;
    _Float16* vdst = &Vls[buf][tid][0];
    #pragma unroll
    for (int i = 0; i < 4; ++i)
      async_ld16(vsrc + i * 8, lds_off(vdst + i * 8));
  };

  issue(0, 0);
  for (int it = 0; it < NBLK; ++it) {
    const int cur = it & 1;
    if (it + 1 < NBLK) { issue(cur ^ 1, (it + 1) * KT); wait_async8(); }
    else               { wait_async0(); }
    __syncthreads();

    // ---- S = Q * K^T : per tile, batch 4 B-fragment loads then 4 WMMAs ----
    v8f s0 = vzero, s1 = vzero;
    #pragma unroll
    for (int t = 0; t < 2; ++t) {
      v16h kf[4];
      #pragma unroll
      for (int ks = 0; ks < 4; ++ks) {
        const _Float16* kr = &Kls[cur][t * 16 + lo16][ks * 32];
        kf[ks] = pack16(*(const v8h*)(kr + hi * 8),
                        *(const v8h*)(kr + 16 + hi * 8));
      }
      v8f s = vzero;
      #pragma unroll
      for (int ks = 0; ks < 4; ++ks)
        s = __builtin_amdgcn_wmma_f32_16x16x32_f16(false, qa[ks], false,
                kf[ks], (short)0, s, false, false);
      if (t == 0) s0 = s; else s1 = s;
    }

    // ---- online softmax: shuffle reduction for row max only ----
    float alpha[8];
    #pragma unroll
    for (int v = 0; v < 8; ++v) {
      float tm = fmaxf(s0[v], s1[v]);
      #pragma unroll
      for (int m = 8; m >= 1; m >>= 1) tm = fmaxf(tm, __shfl_xor(tm, m, 32));
      const float mn = fmaxf(mrow[v], tm);
      alpha[v] = __expf(mrow[v] - mn);
      mrow[v] = mn;
      s0[v] = __expf(s0[v] - mn);
      s1[v] = __expf(s1[v] - mn);
    }
    #pragma unroll
    for (int ct = 0; ct < 8; ++ct)
      #pragma unroll
      for (int v = 0; v < 8; ++v) acc[ct][v] *= alpha[v];
    #pragma unroll
    for (int v = 0; v < 8; ++v) accl[v] *= alpha[v];

    // ---- transpose P (C/D layout -> A layout) through per-wave LDS ----
    #pragma unroll
    for (int v = 0; v < 8; ++v) {
      const int r = v + 8 * hi;
      Pls[wave][r][lo16]      = (_Float16)s0[v];
      Pls[wave][r][16 + lo16] = (_Float16)s1[v];
    }
    const _Float16* pr = &Pls[wave][lo16][0];
    const v16h pa = pack16(*(const v8h*)(pr + hi * 8),
                           *(const v8h*)(pr + 16 + hi * 8));

    // ---- row-sum via matrix pipe: accl += P x ones ----
    accl = __builtin_amdgcn_wmma_f32_16x16x32_f16(false, pa, false, ones,
               (short)0, accl, false, false);

    // ---- O += P * V : batch 4 B-fragments then 4 WMMAs, twice ----
    #pragma unroll
    for (int half = 0; half < 2; ++half) {
      v16h vf[4];
      #pragma unroll
      for (int j = 0; j < 4; ++j) {
        const _Float16* vr = &Vls[cur][(half * 4 + j) * 16 + lo16][0];
        vf[j] = pack16(*(const v8h*)(vr + hi * 8),
                       *(const v8h*)(vr + 16 + hi * 8));
      }
      #pragma unroll
      for (int j = 0; j < 4; ++j) {
        const int ct = half * 4 + j;
        acc[ct] = __builtin_amdgcn_wmma_f32_16x16x32_f16(false, pa, false,
                      vf[j], (short)0, acc[ct], false, false);
      }
    }
    __syncthreads();
  }

  // ---- normalize by row sum (from accl tile) and store E[b][n][c] ----
  #pragma unroll
  for (int ct = 0; ct < 8; ++ct) {
    #pragma unroll
    for (int v = 0; v < 8; ++v) {
      const int r = v + 8 * hi;
      const float val = acc[ct][v] / accl[v];
      const size_t n = (size_t)b * N_ + qbase + wave * 16 + r;
      Eh[n * C_ + ct * 16 + lo16] = (_Float16)val;
    }
  }
}

// ====== Kernel 3: output projection via f32 WMMA + bias + residual ==========
// out[o][hw] = ref + bO[o] + sum_{c<256} wO[o][c] * e_cat[c][hw]
__global__ __launch_bounds__(256) void snl_proj(
    const _Float16* __restrict__ Eh, const float* __restrict__ wO,
    const float* __restrict__ bO, const float* __restrict__ ref,
    float* __restrict__ out)
{
  __shared__ float es[32][2 * C_];   // [token][concat channel], 32 KB
  const int tid = threadIdx.x;
  const int b   = blockIdx.x >> 7;
  const int hw0 = (blockIdx.x & 127) * 32;
  {
    int t = tid >> 3, ch = (tid & 7) * 16;
    const _Float16* r1 = Eh + ((size_t)b * N_ + hw0 + t) * C_ + ch;
    const _Float16* r2 = Eh + ((size_t)b * N_ + HW_ + hw0 + t) * C_ + ch;
    #pragma unroll
    for (int j = 0; j < 16; ++j) es[t][ch + j]      = (float)r1[j];
    #pragma unroll
    for (int j = 0; j < 16; ++j) es[t][C_ + ch + j] = (float)r2[j];
  }
  __syncthreads();

  const int wave = tid >> 5, lane = tid & 31;
  const int lo16 = lane & 15, hi = lane >> 4;

  for (int job = wave; job < 16; job += 8) {
    const int t0 = (job >> 3) * 16;        // hw tile base
    const int og = (job & 7) * 16;         // out-channel tile base
    const float bias = bO[og + lo16];
    v8f acc;
    #pragma unroll
    for (int i = 0; i < 8; ++i) acc[i] = bias;
    const float* arow = &es[t0 + lo16][0];
    const float* brow = wO + (size_t)(og + lo16) * (2 * C_);
    #pragma unroll 4
    for (int ks = 0; ks < 64; ++ks) {
      const int cb = ks * 4 + hi * 2;
      v2f a  = *(const v2f*)(arow + cb);
      v2f bw = *(const v2f*)(brow + cb);
      acc = __builtin_amdgcn_wmma_f32_16x16x4_f32(false, a, false, bw,
                (short)0, acc, false, false);
    }
    #pragma unroll
    for (int v = 0; v < 8; ++v) {
      const int r = v + 8 * hi;
      const size_t idx = ((size_t)(b * C_ + og + lo16)) * HW_ + hw0 + t0 + r;
      out[idx] = ref[idx] + acc[v];
    }
  }
}

extern "C" void kernel_launch(void* const* d_in, const int* in_sizes, int n_in,
                              void* d_out, int out_size, void* d_ws, size_t ws_size,
                              hipStream_t stream) {
  (void)in_sizes; (void)n_in; (void)out_size; (void)ws_size;
  const float* ref = (const float*)d_in[0];
  const float* nei = (const float*)d_in[1];
  const float* wA  = (const float*)d_in[2];
  const float* bA  = (const float*)d_in[3];
  const float* wB  = (const float*)d_in[4];
  const float* bB  = (const float*)d_in[5];
  const float* wD  = (const float*)d_in[6];
  const float* bD  = (const float*)d_in[7];
  const float* wO  = (const float*)d_in[8];
  const float* bO  = (const float*)d_in[9];
  float* out = (float*)d_out;

  _Float16* Qh = (_Float16*)d_ws;                 // [B][N][C]
  _Float16* Kh = Qh + (size_t)B_ * N_ * C_;       // [B][N][C]
  _Float16* Vt = Kh + (size_t)B_ * N_ * C_;       // [B][C][N]
  _Float16* Eh = Vt + (size_t)B_ * N_ * C_;       // [B][N][C]

  snl_qkv<<<B_ * (N_ / 32), 256, 0, stream>>>(ref, nei, wA, bA, wB, bB, wD, bD,
                                              Qh, Kh, Vt);
  snl_attn<<<B_ * (N_ / 64), 128, 0, stream>>>(Qh, Kh, Vt, Eh);
  snl_proj<<<B_ * (HW_ / 32), 256, 0, stream>>>(Eh, wO, bO, ref, out);
}
